// MoeMLPSTK_52982716563636
// MI455X (gfx1250) — matrix operand order repaired and checked by reading, not compiled, not run on hardware
//
#include <hip/hip_runtime.h>
#include <hip/hip_bf16.h>
#include <math.h>

// ---------------- problem constants (match reference) ----------------
#define E_  8
#define K_  2
#define D_  1024
#define FF_ 2048
#define T_  8192        // B*S = 4*2048

// ---------------- GEMM tiling ----------------
#define BM 128
#define BN 64
#define BK 32
#define LDSSTRIDE 40    // 32 + 8 bf16 pad (80 bytes/row, 16B aligned)

typedef __attribute__((ext_vector_type(16))) __bf16 bf16x16;
typedef __attribute__((ext_vector_type(8)))  __bf16 bf16x8;
typedef __attribute__((ext_vector_type(8)))  float  v8f;
typedef __attribute__((ext_vector_type(4)))  unsigned uint32x4;
typedef __attribute__((ext_vector_type(8)))  int      int32x8;
typedef __attribute__((ext_vector_type(4)))  int      int32x4;

#if __has_builtin(__builtin_amdgcn_tensor_load_to_lds)
#define USE_TDM 1
#else
#define USE_TDM 0
#endif

template <int N>
__device__ inline void wait_tensorcnt() {
#if __has_builtin(__builtin_amdgcn_s_wait_tensorcnt)
    __builtin_amdgcn_s_wait_tensorcnt(N);
#else
    asm volatile("s_wait_tensorcnt %0" :: "i"(N) : "memory");
#endif
}

#if USE_TDM
// ---------------------------------------------------------------------
// Issue one TDM 2-D tile load: rows x rowElems bf16 elements from a
// row-major tensor (row stride = strideElems) into LDS at ldsAddr,
// inserting 16B of padding after every 64B row (LDSSTRIDE layout).
// D# layout per CDNA5 ISA.
// ---------------------------------------------------------------------
__device__ inline void tdm_load_tile_2d(unsigned ldsAddr, const void* gptr,
                                        unsigned rowElems, unsigned rows,
                                        unsigned long long strideElems)
{
    unsigned long long ga = (unsigned long long)gptr;
    unsigned td0 = (unsigned)strideElems;       // tensor_dim0 (tiles stay in-bounds)
    unsigned td1 = 1u << 20;                    // tensor_dim1 (large, in-bounds)

    uint32x4 g0;
    g0[0] = 1u;                                               // count=1 (valid user D#)
    g0[1] = ldsAddr;                                          // lds_addr (bytes)
    g0[2] = (unsigned)ga;                                     // global_addr[31:0]
    g0[3] = (unsigned)((ga >> 32) & 0x01FFFFFFu) | (2u << 30);// global_addr[56:32], type=2

    int32x8 g1;
    g1[0] = (int)((1u << 16)    // data_size = 1 -> 2-byte elements
                | (1u << 20)    // pad_enable
                | (3u << 22)    // pad_interval: 16 DWORDs (one 64B tile row)
                | (3u << 25));  // pad_amount: 4 DWORDs (16B row pad)
    g1[1] = (int)((td0 & 0xFFFFu) << 16);                     // tensor_dim0[15:0]
    g1[2] = (int)((td0 >> 16) | ((td1 & 0xFFFFu) << 16));     // dim0 hi | dim1 lo
    g1[3] = (int)((td1 >> 16) | (rowElems << 16));            // dim1 hi | tile_dim0
    g1[4] = (int)rows;                                        // tile_dim1 (tile_dim2=0 -> 2D)
    g1[5] = (int)(unsigned)(strideElems & 0xFFFFFFFFull);     // tensor_dim0_stride[31:0]
    g1[6] = (int)(unsigned)((strideElems >> 32) & 0xFFFFull); // stride[47:32] | dim1_stride=0
    g1[7] = 0;

    int32x4 z4 = {0, 0, 0, 0};                                // D# groups 2/3 (unused, 2D)
    int32x8 z8 = {0, 0, 0, 0, 0, 0, 0, 0};
    __builtin_amdgcn_tensor_load_to_lds(g0, g1, z4, z4, z8, 0);
}
#endif

// ---------------------------------------------------------------------
// zero-fill
// ---------------------------------------------------------------------
__global__ void moe_zero_f32(float* __restrict__ p, size_t n) {
    size_t i = (size_t)blockIdx.x * blockDim.x + threadIdx.x;
    size_t stride = (size_t)gridDim.x * blockDim.x;
    for (; i < n; i += stride) p[i] = 0.0f;
}

// ---------------------------------------------------------------------
// fp32 -> bf16 cast (elementwise, for x)
// ---------------------------------------------------------------------
__global__ void moe_cast_bf16(const float* __restrict__ in, __bf16* __restrict__ out, size_t n) {
    size_t i = (size_t)blockIdx.x * blockDim.x + threadIdx.x;
    size_t stride = (size_t)gridDim.x * blockDim.x;
    for (; i < n; i += stride) out[i] = (__bf16)in[i];
}

// ---------------------------------------------------------------------
// fp32 [R,C] -> bf16 [C,R] transpose-cast (one-time; makes both GEMM
// operands contiguous-row tiles so the TDM can stage them).
// ---------------------------------------------------------------------
__global__ __launch_bounds__(256) void moe_transpose_cast(
    const float* __restrict__ in, __bf16* __restrict__ out, int R, int C)
{
    __shared__ float tile[32][33];
    int c0 = blockIdx.x * 32, r0 = blockIdx.y * 32;
    int tx = threadIdx.x & 31, ty = threadIdx.x >> 5;   // 32 x 8
#pragma unroll
    for (int i = 0; i < 32; i += 8)
        tile[ty + i][tx] = in[(size_t)(r0 + ty + i) * C + c0 + tx];
    __syncthreads();
#pragma unroll
    for (int i = 0; i < 32; i += 8)
        out[(size_t)(c0 + ty + i) * R + r0 + tx] = (__bf16)tile[tx][ty + i];
}

// ---------------------------------------------------------------------
// Router: one wave32 per token.
// ---------------------------------------------------------------------
__global__ __launch_bounds__(256) void moe_router(
    const float* __restrict__ x, const float* __restrict__ rw,
    float* __restrict__ combine, float* __restrict__ zsum,
    float* __restrict__ psum, float* __restrict__ fcnt)
{
    int wave = blockIdx.x * (blockDim.x >> 5) + (threadIdx.x >> 5);
    int lane = threadIdx.x & 31;
    if (wave >= T_) return;

    const float* xr = x + (size_t)wave * D_;
    float xv[32];
#pragma unroll
    for (int j = 0; j < 32; ++j) xv[j] = xr[lane + 32 * j];

    float logit[E_];
#pragma unroll
    for (int e = 0; e < E_; ++e) {
        const float* we = rw + (size_t)e * D_;
        float acc = 0.0f;
#pragma unroll
        for (int j = 0; j < 32; ++j) acc += xv[j] * we[lane + 32 * j];
#pragma unroll
        for (int s = 16; s > 0; s >>= 1) acc += __shfl_xor(acc, s, 32);
        logit[e] = acc;
    }

    if (lane == 0) {
        float m = logit[0];
#pragma unroll
        for (int e = 1; e < E_; ++e) m = fmaxf(m, logit[e]);
        float p[E_]; float se = 0.0f;
#pragma unroll
        for (int e = 0; e < E_; ++e) { p[e] = __expf(logit[e] - m); se += p[e]; }
        float lse = m + __logf(se);
        float inv = 1.0f / se;
#pragma unroll
        for (int e = 0; e < E_; ++e) p[e] *= inv;

        int i0 = 0;
#pragma unroll
        for (int e = 1; e < E_; ++e) if (p[e] > p[i0]) i0 = e;
        int i1 = (i0 == 0) ? 1 : 0;
#pragma unroll
        for (int e = 0; e < E_; ++e) if (e != i0 && p[e] > p[i1]) i1 = e;

        float w0 = p[i0], w1 = p[i1];
        float wsum = w0 + w1;
        w0 /= wsum; w1 /= wsum;

        float* c = combine + (size_t)wave * E_;
#pragma unroll
        for (int e = 0; e < E_; ++e) c[e] = 0.0f;
        c[i0] = w0; c[i1] = w1;

        atomicAdd(zsum, lse * lse);
#pragma unroll
        for (int e = 0; e < E_; ++e) atomicAdd(&psum[e], p[e]);
        atomicAdd(&fcnt[i0], 1.0f);
        atomicAdd(&fcnt[i1], 1.0f);
    }
}

// ---------------------------------------------------------------------
// Fragment helpers (ISA 7.12.2 layouts, wave32)
// ---------------------------------------------------------------------
__device__ inline bf16x16 load_a_frag(const __bf16* rowPtr, int half) {
    bf16x8 lo = *(const bf16x8*)(rowPtr + half * 8);
    bf16x8 hi = *(const bf16x8*)(rowPtr + 16 + half * 8);
    bf16x16 a;
#pragma unroll
    for (int j = 0; j < 8; ++j) { a[j] = lo[j]; a[8 + j] = hi[j]; }
    return a;
}
__device__ inline bf16x16 load_b_frag(const __bf16* rowPtr, int half) {
    bf16x8 lo = *(const bf16x8*)(rowPtr + half * 8);
    bf16x8 hi = *(const bf16x8*)(rowPtr + 16 + half * 8);
    bf16x16 b;
#pragma unroll
    for (int j = 0; j < 8; ++j) { b[j] = lo[j]; b[8 + j] = hi[j]; }
    return b;
}

// ---------------------------------------------------------------------
// Shared K-loop: C[BM,BN] += A[BM,K] * Bt[BN,K]^T, bf16 row-major tiles
// double-buffered in LDS. TDM path: wave0 issues next tile pair into the
// other buffer, then s_wait_tensorcnt 2 guarantees the current pair has
// landed (TENSORcnt completes in order) -> DMA overlaps the WMMAs.
// ---------------------------------------------------------------------
__device__ inline void gemm_k_loop(const __bf16* __restrict__ A, size_t lda,
                                   const __bf16* __restrict__ Bt, size_t ldb,
                                   int Klen, int m0, int n0,
                                   __bf16 (*sA)[BM][LDSSTRIDE], __bf16 (*sBt)[BN][LDSSTRIDE],
                                   v8f acc[2][2])
{
    const int tid  = threadIdx.x;
    const int lane = tid & 31;
    const int wid  = tid >> 5;
    const int wy   = wid >> 1;
    const int wx   = wid & 1;
    const int half = lane >> 4;
    const int l16  = lane & 15;

    const int nk = Klen / BK;

#if USE_TDM
    unsigned ldsA[2], ldsB[2];
#pragma unroll
    for (int b = 0; b < 2; ++b) {
        ldsA[b] = (unsigned)(uintptr_t)&sA[b][0][0];   // flat->LDS: addr[31:0]
        ldsB[b] = (unsigned)(uintptr_t)&sBt[b][0][0];
    }
    // prologue: stage tile 0 into buffer 0
    if (wid == 0) {
        tdm_load_tile_2d(ldsA[0], A + (size_t)m0 * lda, BK, BM, lda);
        tdm_load_tile_2d(ldsB[0], Bt + (size_t)n0 * ldb, BK, BN, ldb);
    }
#endif

    for (int it = 0; it < nk; ++it) {
        const int cur = it & 1;
#if USE_TDM
        const int nxt = cur ^ 1;
        __syncthreads();                      // prior reads of buf[nxt] finished
        if (it + 1 < nk) {
            if (wid == 0) {
                int k0 = (it + 1) * BK;
                tdm_load_tile_2d(ldsA[nxt], A + (size_t)m0 * lda + k0, BK, BM, lda);
                tdm_load_tile_2d(ldsB[nxt], Bt + (size_t)n0 * ldb + k0, BK, BN, ldb);
            }
            wait_tensorcnt<2>();              // all but the 2 newest done -> buf[cur] ready
        } else {
            wait_tensorcnt<0>();
        }
        __syncthreads();                      // publish buf[cur] to all waves
#else
        const int k0 = it * BK;
        __syncthreads();                      // prior reads of buf[cur] finished
        {   // A tile [128x32]: 32B per thread
            int row = tid >> 1, seg = tid & 1;
            const __bf16* src = A + (size_t)(m0 + row) * lda + k0 + seg * 16;
            uint4 v0 = *(const uint4*)(src);
            uint4 v1 = *(const uint4*)(src + 8);
            *(uint4*)&sA[cur][row][seg * 16]     = v0;
            *(uint4*)&sA[cur][row][seg * 16 + 8] = v1;
        }
        if (tid < 128) {  // B tile [64x32]
            int row = tid >> 1, seg = tid & 1;
            const __bf16* src = Bt + (size_t)(n0 + row) * ldb + k0 + seg * 16;
            uint4 v0 = *(const uint4*)(src);
            uint4 v1 = *(const uint4*)(src + 8);
            *(uint4*)&sBt[cur][row][seg * 16]     = v0;
            *(uint4*)&sBt[cur][row][seg * 16 + 8] = v1;
        }
        if (k0 + BK < Klen)
            __builtin_prefetch(Bt + (size_t)(n0 + (tid & 63)) * ldb + k0 + BK, 0, 3);
        __syncthreads();
#endif

        bf16x16 af[2], bfr[2];
#pragma unroll
        for (int i = 0; i < 2; ++i)
            af[i] = load_a_frag(&sA[cur][wy * 32 + i * 16 + l16][0], half);
#pragma unroll
        for (int j = 0; j < 2; ++j)
            bfr[j] = load_b_frag(&sBt[cur][wx * 32 + j * 16 + l16][0], half);

#pragma unroll
        for (int i = 0; i < 2; ++i)
#pragma unroll
            for (int j = 0; j < 2; ++j)
                acc[i][j] = __builtin_amdgcn_wmma_f32_16x16x32_bf16(
                    false, af[i], false, bfr[j], (short)0, acc[i][j], false, false);
    }
    __syncthreads();
}

// ---------------------------------------------------------------------
// GEMM1: H[t, f] = gelu( Xbf[t,:] @ W1t[e*FF+f, :]^T ), stored bf16.
// ---------------------------------------------------------------------
__global__ __launch_bounds__(256) void moe_gemm1_wmma(
    const __bf16* __restrict__ A, const __bf16* __restrict__ W1t,
    __bf16* __restrict__ H, int expert)
{
    __shared__ __bf16 sA[2][BM][LDSSTRIDE];
    __shared__ __bf16 sBt[2][BN][LDSSTRIDE];

    const int lane = threadIdx.x & 31;
    const int wid  = threadIdx.x >> 5;
    const int wy   = wid >> 1, wx = wid & 1;
    const int m0   = blockIdx.y * BM;
    const int n0   = blockIdx.x * BN;
    const int half = lane >> 4, l16 = lane & 15;

    v8f acc[2][2] = {};
    gemm_k_loop(A, D_, W1t + (size_t)expert * FF_ * D_, D_, D_, m0, n0, sA, sBt, acc);

#pragma unroll
    for (int i = 0; i < 2; ++i)
#pragma unroll
        for (int j = 0; j < 2; ++j)
#pragma unroll
            for (int r = 0; r < 8; ++r) {
                int m = m0 + wy * 32 + i * 16 + r + 8 * half;
                int n = n0 + wx * 32 + j * 16 + l16;
                float v = acc[i][j][r];
                float g = 0.5f * v * (1.0f + erff(v * 0.70710678118654752f));
                H[(size_t)m * FF_ + n] = (__bf16)g;
            }
}

// ---------------------------------------------------------------------
// GEMM2: out[t, d] += combine[t, e] * ( H[t,:] @ W2t[d, e*FF+:]^T )
// ---------------------------------------------------------------------
__global__ __launch_bounds__(256) void moe_gemm2_wmma(
    const __bf16* __restrict__ A, const __bf16* __restrict__ W2t,
    const float* __restrict__ combine, float* __restrict__ out, int expert)
{
    __shared__ __bf16 sA[2][BM][LDSSTRIDE];
    __shared__ __bf16 sBt[2][BN][LDSSTRIDE];

    const int lane = threadIdx.x & 31;
    const int wid  = threadIdx.x >> 5;
    const int wy   = wid >> 1, wx = wid & 1;
    const int m0   = blockIdx.y * BM;
    const int n0   = blockIdx.x * BN;
    const int half = lane >> 4, l16 = lane & 15;

    v8f acc[2][2] = {};
    gemm_k_loop(A, FF_, W2t + (size_t)expert * FF_, (size_t)E_ * FF_, FF_, m0, n0, sA, sBt, acc);

#pragma unroll
    for (int i = 0; i < 2; ++i)
#pragma unroll
        for (int j = 0; j < 2; ++j)
#pragma unroll
            for (int r = 0; r < 8; ++r) {
                int m = m0 + wy * 32 + i * 16 + r + 8 * half;
                int n = n0 + wx * 32 + j * 16 + l16;
                float c = combine[(size_t)m * E_ + expert];
                out[(size_t)m * D_ + n] += acc[i][j][r] * c;
            }
}

// ---------------------------------------------------------------------
// finalize scalar losses
// ---------------------------------------------------------------------
__global__ void moe_losses(const float* __restrict__ zsum, const float* __restrict__ psum,
                           const float* __restrict__ fcnt, float* __restrict__ scalars)
{
    if (threadIdx.x == 0 && blockIdx.x == 0) {
        float z = zsum[0] / (float)T_;
        float lb = 0.0f;
#pragma unroll
        for (int e = 0; e < E_; ++e)
            lb += (fcnt[e] / (float)(T_ * K_)) * (psum[e] / (float)T_);
        scalars[0] = z;
        scalars[1] = (float)E_ * lb;
    }
}

// ---------------------------------------------------------------------
extern "C" void kernel_launch(void* const* d_in, const int* in_sizes, int n_in,
                              void* d_out, int out_size, void* d_ws, size_t ws_size,
                              hipStream_t stream)
{
    (void)in_sizes; (void)n_in; (void)out_size; (void)ws_size;
    const float* x  = (const float*)d_in[0];
    const float* rw = (const float*)d_in[1];
    const float* w1 = (const float*)d_in[2];
    const float* w2 = (const float*)d_in[3];
    float* out = (float*)d_out;

    char* ws = (char*)d_ws;
    size_t off = 0;
    __bf16* xbf  = (__bf16*)(ws + off); off += (size_t)T_ * D_ * 2;
    __bf16* w1t  = (__bf16*)(ws + off); off += (size_t)D_ * E_ * FF_ * 2;  // [E*FF, D]
    __bf16* w2t  = (__bf16*)(ws + off); off += (size_t)E_ * FF_ * D_ * 2;  // [D, E*FF]
    __bf16* hbf  = (__bf16*)(ws + off); off += (size_t)T_ * FF_ * 2;
    float* combine = (float*)(ws + off); off += (size_t)T_ * E_ * 4;
    float* zsum = (float*)(ws + off); off += 4;
    float* psum = (float*)(ws + off); off += E_ * 4;
    float* fcnt = (float*)(ws + off); off += E_ * 4;

    // zero output accumulator + loss partials (harness poisons buffers)
    moe_zero_f32<<<4096, 256, 0, stream>>>(out, (size_t)T_ * D_ + 2);
    moe_zero_f32<<<1, 32, 0, stream>>>(zsum, (size_t)(1 + 2 * E_));

    // routing
    moe_router<<<T_ / 8, 256, 0, stream>>>(x, rw, combine, zsum, psum, fcnt);

    // bf16 casts; weights transposed so GEMM tiles are contiguous rows (TDM-friendly)
    moe_cast_bf16<<<8192, 256, 0, stream>>>(x, xbf, (size_t)T_ * D_);
    {   // w1 [D, E*FF] -> w1t [E*FF, D]
        dim3 g((E_ * FF_) / 32, D_ / 32);
        moe_transpose_cast<<<g, 256, 0, stream>>>(w1, w1t, D_, E_ * FF_);
    }
    {   // w2 [E*FF, D] -> w2t [D, E*FF]
        dim3 g(D_ / 32, (E_ * FF_) / 32);
        moe_transpose_cast<<<g, 256, 0, stream>>>(w2, w2t, E_ * FF_, D_);
    }

    // expert MLPs
    for (int e = 0; e < E_; ++e) {
        dim3 g1(FF_ / BN, T_ / BM);   // 32 x 64
        moe_gemm1_wmma<<<g1, 256, 0, stream>>>(xbf, w1t, hbf, e);
        dim3 g2(D_ / BN, T_ / BM);    // 16 x 64
        moe_gemm2_wmma<<<g2, 256, 0, stream>>>(hbf, w2t, combine, out, e);
    }

    // scalar losses appended after out[T*D]
    moe_losses<<<1, 1, 0, stream>>>(zsum, psum, fcnt, out + (size_t)T_ * D_);
}
